// MHA_31739808317641
// MI455X (gfx1250) — compile-verified, matrix operand order checked by reading, not run on hardware
//
#include <hip/hip_runtime.h>

// ---------------------------------------------------------------------------
// MHA forward for MI455X (gfx1250, wave32, WMMA, TDM async tile loads).
// GEMMs: v_wmma_f32_16x16x32_bf16 fed from LDS tiles loaded by the
// Tensor Data Mover (double-buffered, TENSORcnt-synchronized).
// ---------------------------------------------------------------------------

#define B_      4
#define S_      2048
#define DIN_    1024
#define DM_     1024
#define H_      16
#define DH_     64

typedef __attribute__((ext_vector_type(16))) __bf16          v16bf;
typedef __attribute__((ext_vector_type(8)))  float           v8f;
typedef __attribute__((ext_vector_type(16))) unsigned short  v16u;
typedef __attribute__((ext_vector_type(8)))  unsigned short  v8u;
typedef __attribute__((ext_vector_type(4)))  unsigned int    v4ui;
typedef __attribute__((ext_vector_type(8)))  int             v8i;
typedef __attribute__((ext_vector_type(4)))  int             v4i;

#define HAVE_TDM __has_builtin(__builtin_amdgcn_tensor_load_to_lds)

// float -> bf16 (round to nearest even), raw u16
__device__ __forceinline__ unsigned short f2bfu(float f) {
    unsigned u = __builtin_bit_cast(unsigned, f);
    u += 0x7fffu + ((u >> 16) & 1u);
    return (unsigned short)(u >> 16);
}

// 16x32 bf16 fragment load (A-operand layout; also the B-operand when the
// source is stored transposed so lane n reads column n). Two 16-byte loads.
__device__ __forceinline__ v16bf ld_frag(const unsigned short* base, int ld,
                                         int row0, int k0, int lane) {
    int m = lane & 15;
    int h = (lane >> 4) & 1;
    const unsigned short* p = base + (size_t)(row0 + m) * ld + k0 + 8 * h;
    v8u lo = *(const v8u*)p;
    v8u hi = *(const v8u*)(p + 16);
    v16u u;
#pragma unroll
    for (int i = 0; i < 8; ++i) { u[i] = lo[i]; u[i + 8] = hi[i]; }
    return __builtin_bit_cast(v16bf, u);
}

__device__ __forceinline__ v8f wmma_bf16(v16bf a, v16bf b, v8f c) {
    return __builtin_amdgcn_wmma_f32_16x16x32_bf16(
        false, a, false, b, (short)0, c, false, false);
}

// ---------------------------------------------------------------------------
// TDM: async 2-D tile load Global -> LDS (rows x cols bf16, row-major source
// with pitch ld_elems; packed contiguously in LDS with row stride = cols).
// D# packing per CDNA5 ISA.
// ---------------------------------------------------------------------------
#if HAVE_TDM
__device__ __forceinline__ void tdm_load_tile(unsigned lds_off,
                                              const unsigned short* gsrc,
                                              unsigned rows, unsigned cols,
                                              unsigned ld_elems) {
    unsigned long long ga = (unsigned long long)gsrc;

    v4ui g0;
    g0[0] = 1u;                                           // count=1
    g0[1] = lds_off;                                      // lds_addr
    g0[2] = (unsigned)(ga & 0xffffffffu);                 // global_addr[31:0]
    g0[3] = (unsigned)((ga >> 32) & 0x1ffffffu) | 0x80000000u; // addr[56:32] | type=2

    unsigned td0 = ld_elems;        // tensor_dim0 (line length)
    unsigned td1 = 1u << 20;        // tensor_dim1 (large; tiles never OOB)
    unsigned st0 = ld_elems;        // tensor_dim0_stride
    v8i g1;
    g1[0] = (int)(1u << 16);                              // data_size=2B, mask=0
    g1[1] = (int)((td0 & 0xffffu) << 16);                 // tensor_dim0[15:0]
    g1[2] = (int)(((td0 >> 16) & 0xffffu) | ((td1 & 0xffffu) << 16));
    g1[3] = (int)(((td1 >> 16) & 0xffffu) | (cols << 16));   // tile_dim0
    g1[4] = (int)(rows & 0xffffu);                        // tile_dim1 (tile_dim2=0)
    g1[5] = (int)st0;                                     // dim0_stride[31:0]
    g1[6] = 0;                                            // stride[47:32] | dim1_stride lo
    g1[7] = 0;

    v4i z4 = {0, 0, 0, 0};
#if defined(__clang_major__) && __clang_major__ >= 23
    v8i z8 = {0, 0, 0, 0, 0, 0, 0, 0};
    __builtin_amdgcn_tensor_load_to_lds(g0, g1, z4, z4, z8, 0);
#else
    __builtin_amdgcn_tensor_load_to_lds(g0, g1, z4, z4, 0);
#endif
}
#else
// Fallback: cooperative synchronous copy of a 128x32 tile (256 threads).
__device__ __forceinline__ void stage_tile(unsigned short* dst,
                                           const unsigned short* src, int ld) {
    for (int c = threadIdx.x; c < 512; c += 256) {
        int row = c >> 2, k8 = (c & 3) * 8;
        *(v8u*)(dst + row * 32 + k8) = *(const v8u*)(src + (size_t)row * ld + k8);
    }
}
#endif

__device__ __forceinline__ unsigned lds_offset_of(const void* p) {
    return (unsigned)(unsigned long long)p;   // generic ptr low 32 bits = LDS offset
}

// ---------------------------------------------------------------------------
// Kernel 1: fp32 -> bf16 cast
// ---------------------------------------------------------------------------
__global__ __launch_bounds__(256) void k_cast(const float* __restrict__ in,
                                              unsigned short* __restrict__ out,
                                              int n) {
    int stride = gridDim.x * blockDim.x;
    for (int i = blockIdx.x * blockDim.x + threadIdx.x; i < n; i += stride)
        out[i] = f2bfu(in[i]);
}

// ---------------------------------------------------------------------------
// Kernel 2: fp32 [1024,1024] -> bf16 transposed [o,i]
// ---------------------------------------------------------------------------
__global__ __launch_bounds__(256) void k_transpose(const float* __restrict__ w,
                                                   unsigned short* __restrict__ out) {
    int stride = gridDim.x * blockDim.x;
    for (int idx = blockIdx.x * blockDim.x + threadIdx.x; idx < (1 << 20); idx += stride) {
        int o = idx >> 10;
        int i = idx & 1023;
        out[(o << 10) | i] = f2bfu(w[(i << 10) | o]);
    }
}

// ---------------------------------------------------------------------------
// Shared GEMM core: block computes a 128x128 tile of Y = X(8192|..,1024) @ W^T
// tiles staged in LDS by TDM, double-buffered. Each of 8 waves owns 32x64.
// Accumulators returned in c[2][4]; caller stores.
// ---------------------------------------------------------------------------
struct GemmAcc { v8f c[2][4]; };

__device__ __forceinline__ GemmAcc gemm_128x128(const unsigned short* __restrict__ Xb,
                                                const unsigned short* __restrict__ Wt,
                                                int rb, int cb,
                                                unsigned short* sA,   // [2][128*32]
                                                unsigned short* sB) { // [2][128*32]
    int lane = threadIdx.x & 31;
    int wid  = threadIdx.x >> 5;
    int wm   = wid >> 1;            // 0..3 -> row group of 32
    int wn   = wid & 1;             // 0..1 -> col group of 64

    GemmAcc acc;
#pragma unroll
    for (int mi = 0; mi < 2; ++mi)
#pragma unroll
        for (int t = 0; t < 4; ++t) acc.c[mi][t] = (v8f){0, 0, 0, 0, 0, 0, 0, 0};

    const int KSTEPS = DIN_ / 32;

#if HAVE_TDM
    if (wid == 0) {
        tdm_load_tile(lds_offset_of(sA), Xb + (size_t)(rb * 128) * DIN_, 128, 32, DIN_);
        tdm_load_tile(lds_offset_of(sB), Wt + (size_t)(cb * 128) * DIN_, 128, 32, DIN_);
    }
#endif

    for (int i = 0; i < KSTEPS; ++i) {
        int cur = i & 1;
#if HAVE_TDM
        if (wid == 0) {
            if (i + 1 < KSTEPS) {
                int nxt = (i + 1) & 1;
                tdm_load_tile(lds_offset_of(sA + nxt * (128 * 32)),
                              Xb + (size_t)(rb * 128) * DIN_ + (i + 1) * 32, 128, 32, DIN_);
                tdm_load_tile(lds_offset_of(sB + nxt * (128 * 32)),
                              Wt + (size_t)(cb * 128) * DIN_ + (i + 1) * 32, 128, 32, DIN_);
                __builtin_amdgcn_s_wait_tensorcnt(2);  // stage i landed
            } else {
                __builtin_amdgcn_s_wait_tensorcnt(0);
            }
        }
        __syncthreads();
#else
        stage_tile(sA, Xb + (size_t)(rb * 128) * DIN_ + i * 32, DIN_);
        stage_tile(sB, Wt + (size_t)(cb * 128) * DIN_ + i * 32, DIN_);
        cur = 0;
        __syncthreads();
#endif
        const unsigned short* a = sA + cur * (128 * 32);
        const unsigned short* b = sB + cur * (128 * 32);

        v16bf a0 = ld_frag(a, 32, wm * 32,      0, lane);
        v16bf a1 = ld_frag(a, 32, wm * 32 + 16, 0, lane);
#pragma unroll
        for (int t = 0; t < 4; ++t) {
            v16bf bf = ld_frag(b, 32, wn * 64 + t * 16, 0, lane);
            acc.c[0][t] = wmma_bf16(a0, bf, acc.c[0][t]);
            acc.c[1][t] = wmma_bf16(a1, bf, acc.c[1][t]);
        }
        __syncthreads();
    }
    return acc;
}

// ---------------------------------------------------------------------------
// Kernel 3: fused QKV projection (TDM/LDS-tiled GEMM).
// Q,K -> bf16 [B,H,S,64]; V -> transposed bf16 [B,H,64,S].
// ---------------------------------------------------------------------------
__global__ __launch_bounds__(256) void k_qkv(const unsigned short* __restrict__ xbf,
                                             const unsigned short* __restrict__ wqT,
                                             const unsigned short* __restrict__ wkT,
                                             const unsigned short* __restrict__ wvT,
                                             const float* __restrict__ bq,
                                             const float* __restrict__ bk,
                                             const float* __restrict__ bv,
                                             unsigned short* __restrict__ Qo,
                                             unsigned short* __restrict__ Ko,
                                             unsigned short* __restrict__ VTo) {
    __shared__ unsigned short sA[2 * 128 * 32];
    __shared__ unsigned short sB[2 * 128 * 32];

    int bx  = blockIdx.x;
    int mat = bx % 3;                 // 0=Q 1=K 2=V
    int tb  = bx / 3;
    int cb  = tb & 7;                 // 128-col block (1024/128)
    int rb  = tb >> 3;                // 128-row block (8192/128)

    const unsigned short* wT = (mat == 0) ? wqT : (mat == 1) ? wkT : wvT;
    GemmAcc acc = gemm_128x128(xbf, wT, rb, cb, sA, sB);

    int lane = threadIdx.x & 31;
    int n    = lane & 15;
    int hh   = (lane >> 4) & 1;
    int wid  = threadIdx.x >> 5;
    int wm   = wid >> 1;
    int wn   = wid & 1;

    const float* bias = (mat == 0) ? bq : (mat == 1) ? bk : bv;

#pragma unroll
    for (int mi = 0; mi < 2; ++mi)
#pragma unroll
        for (int t = 0; t < 4; ++t) {
            int o    = cb * 128 + wn * 64 + t * 16 + n;
            int head = o >> 6;
            int d    = o & 63;
            float bb = bias[o];
#pragma unroll
            for (int r = 0; r < 8; ++r) {
                int i = rb * 128 + wm * 32 + mi * 16 + r + 8 * hh;  // [0, B*S)
                int b = i >> 11;
                int s = i & (S_ - 1);
                unsigned short v = f2bfu(acc.c[mi][t][r] + bb);
                if (mat == 2)
                    VTo[((size_t)(b * H_ + head) * DH_ + d) * S_ + s] = v;
                else {
                    unsigned short* dst = (mat == 0) ? Qo : Ko;
                    dst[((size_t)(b * H_ + head) * S_ + s) * DH_ + d] = v;
                }
            }
        }
}

// ---------------------------------------------------------------------------
// Kernel 4: causal flash attention. One wave per 16-row Q tile.
// ---------------------------------------------------------------------------
__global__ __launch_bounds__(256) void k_attn(const unsigned short* __restrict__ Q,
                                              const unsigned short* __restrict__ K,
                                              const unsigned short* __restrict__ VT,
                                              unsigned short* __restrict__ Obf) {
    __shared__ unsigned short lds[8 * 16 * 40];

    int wave = blockIdx.x * (blockDim.x >> 5) + (threadIdx.x >> 5);
    int lane = threadIdx.x & 31;
    int n    = lane & 15;
    int hh   = (lane >> 4) & 1;

    int qt = wave & (S_ / 16 - 1);
    int bh = wave >> 7;
    int q0 = qt * 16;

    const unsigned short* Qb = Q  + (size_t)bh * S_ * DH_;
    const unsigned short* Kb = K  + (size_t)bh * S_ * DH_;
    const unsigned short* Vb = VT + (size_t)bh * DH_ * S_;

    v16bf a0 = ld_frag(Qb, DH_, q0, 0,  lane);
    v16bf a1 = ld_frag(Qb, DH_, q0, 32, lane);

    float mrun[8], lrun[8];
    v8f acc[4];
#pragma unroll
    for (int r = 0; r < 8; ++r) { mrun[r] = -3.0e38f; lrun[r] = 0.0f; }
#pragma unroll
    for (int t = 0; t < 4; ++t) acc[t] = (v8f){0, 0, 0, 0, 0, 0, 0, 0};

    unsigned short* pl = lds + (threadIdx.x >> 5) * 16 * 40;

    int nblk = (q0 + 16 + 31) >> 5;
    for (int jb = 0; jb < nblk; ++jb) {
        int j0 = jb * 32;
        if (jb + 1 < nblk) {
            __builtin_prefetch(Kb + (size_t)(j0 + 32 + n) * DH_, 0, 1);
            __builtin_prefetch(Vb + (size_t)n * S_ + j0 + 32, 0, 1);
        }

        v16bf bk00 = ld_frag(Kb, DH_, j0,      0,  lane);
        v16bf bk01 = ld_frag(Kb, DH_, j0,      32, lane);
        v16bf bk10 = ld_frag(Kb, DH_, j0 + 16, 0,  lane);
        v16bf bk11 = ld_frag(Kb, DH_, j0 + 16, 32, lane);

        v8f c0 = (v8f){0, 0, 0, 0, 0, 0, 0, 0};
        v8f c1 = (v8f){0, 0, 0, 0, 0, 0, 0, 0};
        c0 = wmma_bf16(a0, bk00, c0);  c0 = wmma_bf16(a1, bk01, c0);
        c1 = wmma_bf16(a0, bk10, c1);  c1 = wmma_bf16(a1, bk11, c1);

        const float sc = 0.03125f;            // 1/sqrt(D_MODEL)
#pragma unroll
        for (int r = 0; r < 8; ++r) { c0[r] *= sc; c1[r] *= sc; }

        if (j0 + 31 > q0) {
            int k0c = j0 + n, k1c = j0 + 16 + n;
#pragma unroll
            for (int r = 0; r < 8; ++r) {
                int qrow = q0 + r + 8 * hh;
                if (k0c > qrow) c0[r] += -1.0e9f;
                if (k1c > qrow) c1[r] += -1.0e9f;
            }
        }

        float bm[8];
#pragma unroll
        for (int r = 0; r < 8; ++r) bm[r] = fmaxf(c0[r], c1[r]);
        for (int off = 1; off < 16; off <<= 1)
#pragma unroll
            for (int r = 0; r < 8; ++r) bm[r] = fmaxf(bm[r], __shfl_xor(bm[r], off, 16));

        float al[8], ps[8];
#pragma unroll
        for (int r = 0; r < 8; ++r) {
            float nm = fmaxf(mrun[r], bm[r]);
            al[r]   = __expf(mrun[r] - nm);
            mrun[r] = nm;
            c0[r]   = __expf(c0[r] - nm);
            c1[r]   = __expf(c1[r] - nm);
            ps[r]   = c0[r] + c1[r];
        }
        for (int off = 1; off < 16; off <<= 1)
#pragma unroll
            for (int r = 0; r < 8; ++r) ps[r] += __shfl_xor(ps[r], off, 16);
#pragma unroll
        for (int r = 0; r < 8; ++r) lrun[r] = lrun[r] * al[r] + ps[r];
#pragma unroll
        for (int t = 0; t < 4; ++t)
#pragma unroll
            for (int r = 0; r < 8; ++r) acc[t][r] *= al[r];

        // C-layout -> A-layout via per-wave LDS tile (LDS in-order per wave)
#pragma unroll
        for (int r = 0; r < 8; ++r) {
            int row = r + 8 * hh;
            pl[row * 40 + n]      = f2bfu(c0[r]);
            pl[row * 40 + 16 + n] = f2bfu(c1[r]);
        }
        asm volatile("s_wait_dscnt 0" ::: "memory");
        v16bf pa = ld_frag(pl, 40, 0, 0, lane);

#pragma unroll
        for (int t = 0; t < 4; ++t) {
            v16bf bv = ld_frag(Vb, S_, t * 16, j0, lane);
            acc[t] = wmma_bf16(pa, bv, acc[t]);
        }
    }

    float inv[8];
#pragma unroll
    for (int r = 0; r < 8; ++r) inv[r] = 1.0f / lrun[r];

    int bb = bh >> 4;
    int hd = bh & (H_ - 1);
#pragma unroll
    for (int t = 0; t < 4; ++t)
#pragma unroll
        for (int r = 0; r < 8; ++r) {
            int gi  = bb * S_ + q0 + r + 8 * hh;
            int col = hd * DH_ + t * 16 + n;
            Obf[(size_t)gi * DM_ + col] = f2bfu(acc[t][r] * inv[r]);
        }
}

// ---------------------------------------------------------------------------
// Kernel 5: output projection (TDM/LDS-tiled GEMM), f32 result to d_out.
// ---------------------------------------------------------------------------
__global__ __launch_bounds__(256) void k_dense(const unsigned short* __restrict__ abf,
                                               const unsigned short* __restrict__ dT,
                                               const float* __restrict__ db,
                                               float* __restrict__ out) {
    __shared__ unsigned short sA[2 * 128 * 32];
    __shared__ unsigned short sB[2 * 128 * 32];

    int bx = blockIdx.x;
    int cb = bx & 7;
    int rb = bx >> 3;

    GemmAcc acc = gemm_128x128(abf, dT, rb, cb, sA, sB);

    int lane = threadIdx.x & 31;
    int n    = lane & 15;
    int hh   = (lane >> 4) & 1;
    int wid  = threadIdx.x >> 5;
    int wm   = wid >> 1;
    int wn   = wid & 1;

#pragma unroll
    for (int mi = 0; mi < 2; ++mi)
#pragma unroll
        for (int t = 0; t < 4; ++t) {
            int o = cb * 128 + wn * 64 + t * 16 + n;
            float bb = db[o];
#pragma unroll
            for (int r = 0; r < 8; ++r) {
                int i = rb * 128 + wm * 32 + mi * 16 + r + 8 * hh;
                out[(size_t)i * DM_ + o] = acc.c[mi][t][r] + bb;
            }
        }
}

// ---------------------------------------------------------------------------
// Host launcher
// ---------------------------------------------------------------------------
extern "C" void kernel_launch(void* const* d_in, const int* in_sizes, int n_in,
                              void* d_out, int out_size, void* d_ws, size_t ws_size,
                              hipStream_t stream) {
    const float* x  = (const float*)d_in[0];
    // d_in[1] = mask (causal, implemented analytically)
    const float* wq = (const float*)d_in[2];
    const float* bq = (const float*)d_in[3];
    const float* wk = (const float*)d_in[4];
    const float* bk = (const float*)d_in[5];
    const float* wv = (const float*)d_in[6];
    const float* bv = (const float*)d_in[7];
    const float* dw = (const float*)d_in[8];
    const float* db = (const float*)d_in[9];

    char* ws = (char*)d_ws;
    size_t off = 0;
    auto alloc = [&](size_t bytes) {
        char* p = ws + off;
        off = (off + bytes + 255) & ~(size_t)255;
        return p;
    };
    const size_t XB  = (size_t)B_ * S_ * DIN_ * 2;       // 16 MB
    const size_t WB  = (size_t)DIN_ * DM_ * 2;           // 2 MB
    const size_t QKB = (size_t)B_ * H_ * S_ * DH_ * 2;   // 16 MB

    unsigned short* xbf    = (unsigned short*)alloc(XB);
    unsigned short* wqT    = (unsigned short*)alloc(WB);
    unsigned short* wkT    = (unsigned short*)alloc(WB);
    unsigned short* wvT    = (unsigned short*)alloc(WB);
    unsigned short* dTT    = (unsigned short*)alloc(WB);
    unsigned short* Qb     = (unsigned short*)alloc(QKB);
    unsigned short* Kb     = (unsigned short*)alloc(QKB);
    unsigned short* VTb    = (unsigned short*)alloc(QKB);
    unsigned short* attnbf = (unsigned short*)alloc(XB);

    k_cast<<<2048, 256, 0, stream>>>(x, xbf, B_ * S_ * DIN_);
    k_transpose<<<4096, 256, 0, stream>>>(wq, wqT);
    k_transpose<<<4096, 256, 0, stream>>>(wk, wkT);
    k_transpose<<<4096, 256, 0, stream>>>(wv, wvT);
    k_transpose<<<4096, 256, 0, stream>>>(dw, dTT);

    // QKV: 64 row-blocks * 8 col-blocks * 3 mats
    k_qkv<<<64 * 8 * 3, 256, 0, stream>>>(xbf, wqT, wkT, wvT, bq, bk, bv, Qb, Kb, VTb);

    // attention: B*H*(S/16) = 8192 waves / 8 per block
    k_attn<<<1024, 256, 0, stream>>>(Qb, Kb, VTb, attnbf);

    // dense: 64 row-blocks * 8 col-blocks
    k_dense<<<64 * 8, 256, 0, stream>>>(attnbf, dTT, db, (float*)d_out);
}
